// DQA_12799002542473
// MI455X (gfx1250) — matrix operand order repaired and checked by reading, not compiled, not run on hardware
//
#include <hip/hip_runtime.h>
#include <hip/hip_bf16.h>

typedef __attribute__((ext_vector_type(16))) _Float16 v16h;
typedef __attribute__((ext_vector_type(8)))  _Float16 v8h;
typedef __attribute__((ext_vector_type(8)))  float    v8f;
typedef __attribute__((ext_vector_type(4)))  int      v4i;

#define DEV __device__ __forceinline__

#if __has_builtin(__builtin_amdgcn_global_load_async_to_lds_b128) && \
    __has_builtin(__builtin_amdgcn_s_wait_asynccnt)
#define HAVE_ASYNC_LDS 1
typedef __attribute__((address_space(1))) v4i* as1_v4i;
typedef __attribute__((address_space(3))) v4i* as3_v4i;
#else
#define HAVE_ASYNC_LDS 0
#endif

constexpr int kNSrc  = 900;
constexpr int kNTgt  = 262144;
constexpr int kNEdge = 500000;
constexpr int kD     = 128;
constexpr int kH     = 8;

// padded LDS row strides (halves) to break 256B/1024B bank-conflict periodicity;
// 136*2=272B and 520*2=1040B are 16B-aligned and shift banks by 4 per row.
constexpr int kPadD = 136;   // for 128-wide f16 tiles
constexpr int kPadH = 520;   // for 512-wide f16 tiles

// ---------------- workspace layout (bytes) ----------------
constexpr size_t OFF_Q     = 0;                                   // f16  NTGT*128
constexpr size_t OFF_AGG   = OFF_Q     + (size_t)kNTgt * kD * 2;  // f32  NTGT*128
constexpr size_t OFF_MENC  = OFF_AGG   + (size_t)kNTgt * kD * 4;  // u32  NTGT*8
constexpr size_t OFF_DEN   = OFF_MENC  + (size_t)kNTgt * kH * 4;  // f32  NTGT*8
constexpr size_t OFF_ALPHA = OFF_DEN   + (size_t)kNTgt * kH * 4;  // f32  NEDGE*8
constexpr size_t OFF_KH    = OFF_ALPHA + (size_t)kNEdge * kH * 4; // f16  900*128
constexpr size_t OFF_VF    = OFF_KH    + (size_t)kNSrc * kD * 2;  // f32  900*128
constexpr size_t OFF_MASK  = OFF_VF    + (size_t)kNSrc * kD * 4;  // i32  900 (pad 4K)
constexpr size_t OFF_WQT   = OFF_MASK  + 4096;                    // f16  128*128 each
constexpr size_t OFF_WIHT  = OFF_WQT   + 32768;
constexpr size_t OFF_WHHT  = OFF_WIHT  + 32768;
constexpr size_t OFF_WOT   = OFF_WHHT  + 32768;
constexpr size_t OFF_WM1T  = OFF_WOT   + 32768;                   // f16  512*128
constexpr size_t OFF_WM2T  = OFF_WM1T  + 131072;                  // f16  128*512
constexpr size_t OFF_BG    = OFF_WM2T  + 131072;                  // f32  128

// ---------------- helpers ----------------
DEV v8f wmma16(v16h a, v16h b, v8f c) {
  return __builtin_amdgcn_wmma_f32_16x16x32_f16(false, a, false, b, (short)0, c,
                                                false, false);
}

// A operand (16x32 f16 tile): lane holds row (lane&15); halves e=0..7 -> K=kbase+8*hi+e,
// e=8..15 -> K=kbase+16+8*hi+(e-8).  Two 16B chunks.
DEV v16h load_a16(const _Float16* rowM, int kbase, int hi) {
  v8h a0 = *(const v8h*)(rowM + kbase + 8 * hi);
  v8h a1 = *(const v8h*)(rowM + kbase + 16 + 8 * hi);
  return __builtin_shufflevector(a0, a1, 0, 1, 2, 3, 4, 5, 6, 7,
                                 8, 9, 10, 11, 12, 13, 14, 15);
}

// B operand (32x16 f16 tile, from W^T rows): lane holds col (lane&15);
// halves e=0..7 -> K=kbase+8*hi+e, e=8..15 -> K=kbase+16+8*hi+(e-8).  Two 16B chunks.
DEV v16h load_b16(const _Float16* rowN, int kbase, int hi) {
  v8h b0 = *(const v8h*)(rowN + kbase + 8 * hi);
  v8h b1 = *(const v8h*)(rowN + kbase + 16 + 8 * hi);
  return __builtin_shufflevector(b0, b1, 0, 1, 2, 3, 4, 5, 6, 7,
                                 8, 9, 10, 11, 12, 13, 14, 15);
}

// A operand assembled directly from fp32 global rows (convert to f16 in-register).
DEV v16h cvt_a_f32(const float* rowM, int kbase, int hi) {
  float4 c0 = *(const float4*)(rowM + kbase + 8 * hi);
  float4 c1 = *(const float4*)(rowM + kbase + 8 * hi + 4);
  float4 c2 = *(const float4*)(rowM + kbase + 16 + 8 * hi);
  float4 c3 = *(const float4*)(rowM + kbase + 16 + 8 * hi + 4);
  v16h r;
  r[0]  = (_Float16)c0.x; r[1]  = (_Float16)c0.y; r[2]  = (_Float16)c0.z; r[3]  = (_Float16)c0.w;
  r[4]  = (_Float16)c1.x; r[5]  = (_Float16)c1.y; r[6]  = (_Float16)c1.z; r[7]  = (_Float16)c1.w;
  r[8]  = (_Float16)c2.x; r[9]  = (_Float16)c2.y; r[10] = (_Float16)c2.z; r[11] = (_Float16)c2.w;
  r[12] = (_Float16)c3.x; r[13] = (_Float16)c3.y; r[14] = (_Float16)c3.z; r[15] = (_Float16)c3.w;
  return r;
}

// monotonic float<->uint order-preserving encoding for atomic segment-max
DEV unsigned encf(float f) {
  int i = __float_as_int(f);
  return (i >= 0) ? ((unsigned)i ^ 0x80000000u) : ~(unsigned)i;
}
DEV float decf(unsigned u) {
  return (u & 0x80000000u) ? __int_as_float((int)(u ^ 0x80000000u))
                           : __int_as_float((int)~u);
}

// ---------------- kernel 1: init + weight transpose/convert ----------------
__global__ __launch_bounds__(256) void k_prep(
    float* agg, float* den, unsigned* menc,
    _Float16* WqT, _Float16* WihT, _Float16* WhhT, _Float16* WoT,
    _Float16* Wm1T, _Float16* Wm2T, float* bg,
    const float* Wq, const float* Wih, const float* Whh, const float* Wo,
    const float* Wm1, const float* Wm2, const float* bih, const float* bhh) {
  int tid = blockIdx.x * 256 + threadIdx.x;
  if (tid < kNTgt * kD) agg[tid] = 0.f;
  if (tid < kNTgt * kH) { den[tid] = 0.f; menc[tid] = 0u; }
  if (tid < kD * kD) {
    int n = tid >> 7, k = tid & 127;
    int s = k * kD + n;
    WqT[tid]  = (_Float16)Wq[s];
    WihT[tid] = (_Float16)Wih[s];
    WhhT[tid] = (_Float16)Whh[s];
    WoT[tid]  = (_Float16)Wo[s];
  }
  if (tid < 512 * kD) {
    { int n = tid >> 7, k = tid & 127; Wm1T[tid] = (_Float16)Wm1[k * 512 + n]; }
    { int n = tid >> 9, k = tid & 511; Wm2T[tid] = (_Float16)Wm2[k * kD + n]; }
  }
  if (tid < kD) bg[tid] = bih[tid] + bhh[tid];
}

// ---------------- kernel 2: source LN + K/V projections + mask (tiny) ----------------
__global__ __launch_bounds__(128) void k_src(
    const float* xsrc, const float* g1, const float* be1,
    const float* Wk, const float* bk, const float* Wv, const float* bv,
    _Float16* kh, float* vf, int* mask) {
  __shared__ float xs[kD];
  __shared__ float sc[kD];
  int t = threadIdx.x, row = blockIdx.x;
  float v = xsrc[row * kD + t];
  sc[t] = v; __syncthreads();
  for (int off = 64; off > 0; off >>= 1) { if (t < off) sc[t] += sc[t + off]; __syncthreads(); }
  float mu = sc[0] * (1.f / kD);
  __syncthreads();
  float d = v - mu;
  sc[t] = d * d; __syncthreads();
  for (int off = 64; off > 0; off >>= 1) { if (t < off) sc[t] += sc[t + off]; __syncthreads(); }
  float rstd = rsqrtf(sc[0] * (1.f / kD) + 1e-5f);
  __syncthreads();
  float xv = d * rstd * g1[t] + be1[t];
  xs[t] = xv; sc[t] = xv; __syncthreads();
  for (int off = 64; off > 0; off >>= 1) { if (t < off) sc[t] += sc[t + off]; __syncthreads(); }
  float rowsum = sc[0];
  float ka = bk[t], va = bv[t];
  for (int k = 0; k < kD; k++) {
    float x = xs[k];
    ka += x * Wk[k * kD + t];
    va += x * Wv[k * kD + t];
  }
  kh[row * kD + t] = (_Float16)ka;
  vf[row * kD + t] = va;
  if (t == 0) mask[row] = (rowsum != 0.f) ? 1 : 0;
}

// ---------------- kernel 3: Q projection, WMMA f16 ----------------
__global__ __launch_bounds__(256) void k_qproj(
    const float* xt, const float* bq, const _Float16* WqT, _Float16* qh) {
  int lane = threadIdx.x & 31, wave = threadIdx.x >> 5;
  int hi = lane >> 4, lrow = lane & 15;
  int r0 = blockIdx.x * 128 + wave * 16;
  const float* arow = xt + (size_t)(r0 + lrow) * kD;
  v16h A[4];
#pragma unroll
  for (int kt = 0; kt < 4; kt++) A[kt] = cvt_a_f32(arow, kt * 32, hi);
#pragma unroll
  for (int nt = 0; nt < 8; nt++) {
    v8f acc = {};
    int n = nt * 16 + lrow;
    const _Float16* brow = WqT + n * kD;
#pragma unroll
    for (int kt = 0; kt < 4; kt++) acc = wmma16(A[kt], load_b16(brow, kt * 32, hi), acc);
    float bb = bq[n];
#pragma unroll
    for (int i = 0; i < 8; i++)
      qh[(size_t)(r0 + i + 8 * hi) * kD + n] = (_Float16)(acc[i] + bb);
  }
}

// ---------------- kernel 4: per-edge attention logits + segment max ----------------
__global__ __launch_bounds__(256) void k_edge_alpha(
    const int* esrc, const int* edst, const _Float16* qh, const _Float16* kh,
    float* alpha, unsigned* menc) {
  int e = blockIdx.x * 256 + threadIdx.x;
  if (e >= kNEdge) return;
  int s = esrc[e], dgt = edst[e];
  const _Float16* qp = qh + (size_t)dgt * kD;
  const _Float16* kp = kh + (size_t)s * kD;
#pragma unroll
  for (int h = 0; h < kH; h++) {
    v8h q0 = *(const v8h*)(qp + h * 16), q1 = *(const v8h*)(qp + h * 16 + 8);
    v8h k0 = *(const v8h*)(kp + h * 16), k1 = *(const v8h*)(kp + h * 16 + 8);
    float dot = 0.f;
#pragma unroll
    for (int i = 0; i < 8; i++)
      dot += (float)q0[i] * (float)k0[i] + (float)q1[i] * (float)k1[i];
    float a = dot * 0.25f;  // 1/sqrt(16)
    alpha[(size_t)e * kH + h] = a;
    atomicMax(&menc[(size_t)dgt * kH + h], encf(a));
  }
}

// ---------------- kernel 5: exp + denom + unnormalized scatter-accumulate ----------------
__global__ __launch_bounds__(256) void k_edge_accum(
    const int* esrc, const int* edst, const float* alpha, const unsigned* menc,
    const float* vf, const int* mask, float* den, float* agg) {
  long long tid = (long long)blockIdx.x * 256 + threadIdx.x;
  if (tid >= (long long)kNEdge * kH) return;
  int e = (int)(tid >> 3), h = (int)(tid & 7);
  int s = esrc[e], dgt = edst[e];
  float m  = decf(menc[(size_t)dgt * kH + h]);
  float ex = expf(alpha[(size_t)e * kH + h] - m);
  atomicAdd(&den[(size_t)dgt * kH + h], ex);
  if (mask[s]) {
    const float* vp = vf + (size_t)s * kD + h * 16;
    float* ap = agg + (size_t)dgt * kD + h * 16;
#pragma unroll
    for (int d = 0; d < 16; d++) atomicAdd(ap + d, ex * vp[d]);
  }
}

// ---------------- kernel 6: fused gate GEMMs + Wo + residual (WMMA) ----------------
__global__ __launch_bounds__(256) void k_gru(
    const float* xt, const float* agg, const float* den,
    const _Float16* WihT, const _Float16* WhhT, const _Float16* WoT,
    const float* bg, const float* bo, float* out) {
  extern __shared__ _Float16 sm[];  // per wave: aggN/xN/upd, 16 x kPadD f16 each
  int lane = threadIdx.x & 31, wave = threadIdx.x >> 5;
  int hi = lane >> 4, lrow = lane & 15;
  int r0 = blockIdx.x * 128 + wave * 16;
  _Float16* aggN = sm + wave * (3 * 16 * kPadD);
  _Float16* xN   = aggN + 16 * kPadD;
  _Float16* updb = xN + 16 * kPadD;
  for (int rr = 0; rr < 16; rr++) {
    int row = r0 + rr;
    for (int c = lane; c < kD; c += 32) {
      float dn = den[(size_t)row * kH + (c >> 4)] + 1e-16f;
      aggN[rr * kPadD + c] = (_Float16)(agg[(size_t)row * kD + c] / dn);
      xN[rr * kPadD + c]   = (_Float16)(xt[(size_t)row * kD + c]);
    }
  }
  __syncthreads();
  const _Float16* aRow = aggN + lrow * kPadD;
  const _Float16* xRow = xN + lrow * kPadD;
#pragma unroll
  for (int nt = 0; nt < 8; nt++) {
    v8f acc = {};
    int n = nt * 16 + lrow;
    const _Float16* b1 = WihT + n * kD;
    const _Float16* b2 = WhhT + n * kD;
#pragma unroll
    for (int kt = 0; kt < 4; kt++) acc = wmma16(load_a16(aRow, kt * 32, hi), load_b16(b1, kt * 32, hi), acc);
#pragma unroll
    for (int kt = 0; kt < 4; kt++) acc = wmma16(load_a16(xRow, kt * 32, hi), load_b16(b2, kt * 32, hi), acc);
    float bb = bg[n];
#pragma unroll
    for (int i = 0; i < 8; i++) {
      float z = acc[i] + bb;
      float gsig = 1.f / (1.f + expf(-z));
      float av = (float)aggN[(i + 8 * hi) * kPadD + n];
      updb[(i + 8 * hi) * kPadD + n] = (_Float16)(av * gsig);
    }
  }
  __syncthreads();
  const _Float16* uRow = updb + lrow * kPadD;
#pragma unroll
  for (int nt = 0; nt < 8; nt++) {
    v8f acc = {};
    int n = nt * 16 + lrow;
    const _Float16* b3 = WoT + n * kD;
#pragma unroll
    for (int kt = 0; kt < 4; kt++) acc = wmma16(load_a16(uRow, kt * 32, hi), load_b16(b3, kt * 32, hi), acc);
    float bb = bo[n];
#pragma unroll
    for (int i = 0; i < 8; i++) {
      size_t row = (size_t)(r0 + i + 8 * hi);
      out[row * kD + n] = xt[row * kD + n] + acc[i] + bb;
    }
  }
}

// ---------------- kernel 7: fused LN + MLP (512 hidden stays in LDS) ----------------
__global__ __launch_bounds__(128) void k_mlp(
    const float* g3, const float* be3, const _Float16* Wm1T, const _Float16* Wm2T,
    const float* bm1, const float* bm2, float* out) {
  extern __shared__ char smraw[];
  float*    xtile = (float*)smraw;                                   // 64x128 f32
  _Float16* hN    = (_Float16*)(smraw + 64 * kD * 4);                // 64 x kPadD f16
  _Float16* hid   = (_Float16*)(smraw + 64 * kD * 4 + 64 * kPadD * 2); // 64 x kPadH f16
  int tid = threadIdx.x;
  int lane = tid & 31, wave = tid >> 5;
  int hi = lane >> 4, lrow = lane & 15;
  size_t r0 = (size_t)blockIdx.x * 64;
  const float4* src4 = (const float4*)(out + r0 * kD);
  float4* dst4 = (float4*)xtile;
#if HAVE_ASYNC_LDS
  for (int i = tid; i < 64 * kD / 4; i += 128)
    __builtin_amdgcn_global_load_async_to_lds_b128((as1_v4i)(src4 + i),
                                                   (as3_v4i)(dst4 + i), 0, 0);
  __builtin_amdgcn_s_wait_asynccnt(0);
#else
  for (int i = tid; i < 64 * kD / 4; i += 128) dst4[i] = src4[i];
#endif
  __syncthreads();
  if (tid < 64) {
    const float* rw = xtile + tid * kD;
    float su = 0.f;
    for (int c = 0; c < kD; c++) su += rw[c];
    float mu = su * (1.f / kD);
    float vs = 0.f;
    for (int c = 0; c < kD; c++) { float dd = rw[c] - mu; vs += dd * dd; }
    float rstd = rsqrtf(vs * (1.f / kD) + 1e-5f);
    for (int c = 0; c < kD; c++)
      hN[tid * kPadD + c] = (_Float16)((rw[c] - mu) * rstd * g3[c] + be3[c]);
  }
  __syncthreads();
  const _Float16* aRow = hN + (wave * 16 + lrow) * kPadD;
  for (int nt = 0; nt < 32; nt++) {
    v8f acc = {};
    int n = nt * 16 + lrow;
    const _Float16* brow = Wm1T + n * kD;
#pragma unroll
    for (int kt = 0; kt < 4; kt++) acc = wmma16(load_a16(aRow, kt * 32, hi), load_b16(brow, kt * 32, hi), acc);
    float bb = bm1[n];
#pragma unroll
    for (int i = 0; i < 8; i++) {
      float hv = acc[i] + bb;
      hv = hv > 0.f ? hv : 0.f;
      hid[(wave * 16 + i + 8 * hi) * kPadH + n] = (_Float16)hv;
    }
  }
  __syncthreads();
  const _Float16* aRow2 = hid + (wave * 16 + lrow) * kPadH;
#pragma unroll
  for (int nt = 0; nt < 8; nt++) {
    v8f acc = {};
    int n = nt * 16 + lrow;
    const _Float16* brow = Wm2T + n * 512;
#pragma unroll
    for (int kt = 0; kt < 16; kt++) acc = wmma16(load_a16(aRow2, kt * 32, hi), load_b16(brow, kt * 32, hi), acc);
    float bb = bm2[n];
#pragma unroll
    for (int i = 0; i < 8; i++) {
      int lr = wave * 16 + i + 8 * hi;
      out[(r0 + lr) * kD + n] = xtile[lr * kD + n] + acc[i] + bb;
    }
  }
}

// ---------------- launcher ----------------
extern "C" void kernel_launch(void* const* d_in, const int* in_sizes, int n_in,
                              void* d_out, int out_size, void* d_ws, size_t ws_size,
                              hipStream_t stream) {
  (void)in_sizes; (void)n_in; (void)out_size; (void)ws_size;
  const float* x_source = (const float*)d_in[0];
  const float* x_target = (const float*)d_in[1];
  const int*   edge_src = (const int*)d_in[2];
  const int*   edge_dst = (const int*)d_in[3];
  const float* Wq  = (const float*)d_in[4],  *bq  = (const float*)d_in[5];
  const float* Wk  = (const float*)d_in[6],  *bk  = (const float*)d_in[7];
  const float* Wv  = (const float*)d_in[8],  *bv  = (const float*)d_in[9];
  const float* Wih = (const float*)d_in[10], *bih = (const float*)d_in[11];
  const float* Whh = (const float*)d_in[12], *bhh = (const float*)d_in[13];
  const float* Wo  = (const float*)d_in[14], *bo  = (const float*)d_in[15];
  const float* g1  = (const float*)d_in[16], *be1 = (const float*)d_in[17];
  const float* g3  = (const float*)d_in[18], *be3 = (const float*)d_in[19];
  const float* Wm1 = (const float*)d_in[20], *bm1 = (const float*)d_in[21];
  const float* Wm2 = (const float*)d_in[22], *bm2 = (const float*)d_in[23];

  char* ws = (char*)d_ws;
  _Float16* qh    = (_Float16*)(ws + OFF_Q);
  float*    agg   = (float*)(ws + OFF_AGG);
  unsigned* menc  = (unsigned*)(ws + OFF_MENC);
  float*    den   = (float*)(ws + OFF_DEN);
  float*    alpha = (float*)(ws + OFF_ALPHA);
  _Float16* kh    = (_Float16*)(ws + OFF_KH);
  float*    vf    = (float*)(ws + OFF_VF);
  int*      mask  = (int*)(ws + OFF_MASK);
  _Float16* WqT   = (_Float16*)(ws + OFF_WQT);
  _Float16* WihT  = (_Float16*)(ws + OFF_WIHT);
  _Float16* WhhT  = (_Float16*)(ws + OFF_WHHT);
  _Float16* WoT   = (_Float16*)(ws + OFF_WOT);
  _Float16* Wm1T  = (_Float16*)(ws + OFF_WM1T);
  _Float16* Wm2T  = (_Float16*)(ws + OFF_WM2T);
  float*    bg    = (float*)(ws + OFF_BG);
  float*    out   = (float*)d_out;

  size_t lds_gru = (size_t)8 * 3 * 16 * kPadD * 2;                       // 104448 B
  size_t lds_mlp = (size_t)64 * kD * 4 + 64 * kPadD * 2 + 64 * kPadH * 2; // 116736 B

  k_prep<<<(kNTgt * kD) / 256, 256, 0, stream>>>(
      agg, den, menc, WqT, WihT, WhhT, WoT, Wm1T, Wm2T, bg,
      Wq, Wih, Whh, Wo, Wm1, Wm2, bih, bhh);
  k_src<<<kNSrc, 128, 0, stream>>>(x_source, g1, be1, Wk, bk, Wv, bv, kh, vf, mask);
  k_qproj<<<kNTgt / 128, 256, 0, stream>>>(x_target, bq, WqT, qh);
  k_edge_alpha<<<(kNEdge + 255) / 256, 256, 0, stream>>>(edge_src, edge_dst, qh, kh, alpha, menc);
  k_edge_accum<<<(kNEdge * kH + 255) / 256, 256, 0, stream>>>(edge_src, edge_dst, alpha, menc,
                                                              vf, mask, den, agg);
  k_gru<<<kNTgt / 128, 256, lds_gru, stream>>>(x_target, agg, den, WihT, WhhT, WoT, bg, bo, out);
  k_mlp<<<kNTgt / 64, 128, lds_mlp, stream>>>(g3, be3, Wm1T, Wm2T, bm1, bm2, out);
}
